// FillPolygon_79869211837094
// MI455X (gfx1250) — matrix operand order repaired and checked by reading, not compiled, not run on hardware
//
#include <hip/hip_runtime.h>
#include <hip/hip_bf16.h>
#include <math.h>

typedef __attribute__((ext_vector_type(16))) _Float16 v16h;
typedef __attribute__((ext_vector_type(8)))  float    v8f;

#define B_   8
#define H_   256
#define W_   256
#define NV_  64
#define N3_  192   // 3 * NV (wrap-extended table)
#define TWO_PI_ 6.28318530717958647692f
// -100 * log2(e): exp(-100*d2) * 2^14 == exp2(d2 * NEG100_LOG2E + 14)
#define NEG100_LOG2E_ (-144.26950408889634f)

// One wave computes a 16-pixel tile.
// A (16x32 f16): row0 = r1[k], row1 = 1.0, rows 2..15 = 0   (table values)
// B (32x16 f16): B[k][n] = unnormalized RBF weight of pixel n vs table entry k
// D = A*B + C accumulated in f32 over 6 K-chunks (192 table entries):
//   D[0][n] = sum_k w*r   -> lane n, c[0]
//   D[1][n] = sum_k w     -> lane n, c[1]
// Each of lanes 0..15 owns its pixel's result pair directly - no extraction.
__global__ __launch_bounds__(256) void fillpoly_wmma_kernel(
        const float* __restrict__ in, float* __restrict__ out)
{
    __shared__ float s_tt[N3_];
    __shared__ float s_r[N3_];

    const int b      = blockIdx.x >> 9;      // 512 blocks per batch image
    const int blkInB = blockIdx.x & 511;
    const int tid    = threadIdx.x;

    // Build wrap-extended vertex polar tables once per block.
    if (tid < N3_) {
        const int v      = tid & 63;
        const int period = tid >> 6;         // 0,1,2 -> -2pi, 0, +2pi
        const float x = in[(b * NV_ + v) * 2 + 0] - 128.0f;
        const float y = in[(b * NV_ + v) * 2 + 1] - 128.0f;
        const float r = sqrtf(x * x + y * y);
        float t = atan2f(y, x);
        t = (t < 0.0f) ? (t + TWO_PI_) : t;  // fix_radians
        s_tt[tid] = t + TWO_PI_ * (float)(period - 1);
        s_r[tid]  = r;
    }
    __syncthreads();

    const int lane = tid & 31;
    const int wave = tid >> 5;
    const int half = lane >> 4;              // which 16-lane group
    const int n    = lane & 15;              // pixel column N / A-row M

    // Branch-free A-row masks: row 0 carries r1, row 1 carries ones.
    const float row0 = (n == 0) ? 1.0f : 0.0f;
    const float row1 = (n == 1) ? 1.0f : 0.0f;

    // Pixel polar coordinates (lanes L and L+16 handle the same pixel,
    // covering the two K-halves of the B tile).
    const int   p   = blkInB * 128 + wave * 16 + n;   // flat pixel in [0, 65536)
    const int   px  = p & (W_ - 1);
    const int   py  = p >> 8;
    const float gx  = (float)px - 128.0f;
    const float gy  = (float)py - 128.0f;
    const float r_mat = sqrtf(gx * gx + gy * gy);
    float tmat = atan2f(gy, gx);
    tmat = (tmat < 0.0f) ? (tmat + TWO_PI_) : tmat;

    v8f c = {};
    #pragma unroll
    for (int chunk = 0; chunk < 6; ++chunk) {
        const int kbase = chunk * 32;

        // A tile: 16x32 f16. lane -> row M = n; element e -> K per ISA layout.
        // Unconditional load + masked FMA so clang cannot sink the ds_load
        // into divergent saveexec blocks.
        v16h am;
        #pragma unroll
        for (int e = 0; e < 16; ++e) {
            const int k = e + ((e >= 8) ? 8 : 0) + half * 8;
            const float rv = s_r[kbase + k];
            const float av = fmaf(rv, row0, row1);
            am[e] = (_Float16)av;
        }

        // B tile: 32x16 f16 weights. lane -> col N = n; element e -> K per ISA layout (same as A).
        // Whole weight evaluation folded to: sub, sq, min, fma, v_exp_f32, cvt.
        v16h bw;
        #pragma unroll
        for (int e = 0; e < 16; ++e) {
            const float d  = s_tt[kbase + e + ((e >= 8) ? 8 : 0) + half * 8] - tmat;
            const float d2 = fminf(d * d, 1.0f);
            // exp(-100*d2) * 2^14 in one exp2: scale keeps f16 range, cancels in ratio
            const float w  = __builtin_amdgcn_exp2f(fmaf(d2, NEG100_LOG2E_, 14.0f));
            bw[e] = (_Float16)w;
        }

        c = __builtin_amdgcn_wmma_f32_16x16x32_f16(
                /*neg_a=*/false, am, /*neg_b=*/false, bw,
                /*c_mod=*/(short)0, c, /*reuse_a=*/false, /*reuse_b=*/false);
    }

    // Lanes 0..15: c[0] = D[0][n] = sum(w*r), c[1] = D[1][n] = sum(w).
    if (half == 0) {
        const float swr = c[0];
        const float sw  = c[1];
        const float r_rbf = swr / (sw + 1e-20f);
        float flag = 1.0f + r_rbf - r_mat;
        flag = fminf(fmaxf(flag, 0.0f), 1.0f);
        out[b * (H_ * W_) + p] = flag;
    }
}

extern "C" void kernel_launch(void* const* d_in, const int* in_sizes, int n_in,
                              void* d_out, int out_size, void* d_ws, size_t ws_size,
                              hipStream_t stream) {
    (void)in_sizes; (void)n_in; (void)d_ws; (void)ws_size; (void)out_size;
    const float* in  = (const float*)d_in[0];
    float*       out = (float*)d_out;
    // 8 batches * 512 blocks/batch; 256 threads = 8 wave32, 16 pixels/wave.
    fillpoly_wmma_kernel<<<dim3(B_ * 512), dim3(256), 0, stream>>>(in, out);
}
